// PointNet_20461224198193
// MI455X (gfx1250) — matrix operand order repaired and checked by reading, not compiled, not run on hardware
//
#include <hip/hip_runtime.h>
#include <hip/hip_bf16.h>

// ---------------------------------------------------------------------------
// PointNet++ SA layer for MI455X (gfx1250, wave32, WMMA + async-to-LDS).
// GEMM work (ball-query distance matrix, 1x1 conv layers) uses
// v_wmma_f32_16x16x32_f16 (K=32 == FEAT -> 1 WMMA/tile, K=64 -> 2).
// Conv layers stage activation A-blocks into LDS with
// global_load_async_to_lds_b128 (ASYNCcnt) so the 4 N-tile waves sharing an
// M-tile read activations from LDS instead of 4x from L2.
// FPS is a latency-bound sequential argmax chain -> one 1024-thread block per
// batch, register-resident distances, shfl_xor wave32 reductions.
// ---------------------------------------------------------------------------

typedef __attribute__((ext_vector_type(16))) _Float16 v16h;
typedef __attribute__((ext_vector_type(8)))  _Float16 v8h;
typedef __attribute__((ext_vector_type(8)))  float    v8f;

#define BATCH 8
#define NPTS  16384
#define QPTS  512
#define FEAT  32
#define NS1   32
#define NS2   64
#define R1SQ  36.0f
#define R2SQ  64.0f

__device__ inline v16h pack16(v8h lo, v8h hi) {
    v16h r;
#pragma unroll
    for (int i = 0; i < 8; ++i) { r[i] = lo[i]; r[8 + i] = hi[i]; }
    return r;
}

// ---------------------------------------------------------------------------
// Farthest point sampling: 1 block per batch, 1024 threads, 16 pts/thread.
// ---------------------------------------------------------------------------
__global__ __launch_bounds__(1024) void fps_kernel(const float* __restrict__ pc,
                                                   int* __restrict__ sidx) {
    const int b = blockIdx.x;
    const int t = threadIdx.x;
    const int lane = t & 31, wid = t >> 5;

    float px[16], py[16], pz[16], d[16];
    const size_t base0 = (size_t)b * NPTS * 35;
    float x0 = pc[base0 + 0], y0 = pc[base0 + 1], z0 = pc[base0 + 2];
#pragma unroll
    for (int i = 0; i < 16; ++i) {
        int j = i * 1024 + t;
        const float* p = pc + base0 + (size_t)j * 35;
        px[i] = p[0]; py[i] = p[1]; pz[i] = p[2];
        float dx = px[i] - x0, dy = py[i] - y0, dz = pz[i] - z0;
        d[i] = dx * dx + dy * dy + dz * dz;
    }
    if (t == 0) sidx[b * QPTS] = 0;

    __shared__ float sval[32];
    __shared__ int   sjdx[32];
    __shared__ int   swin;

    for (int s = 1; s < QPTS; ++s) {
        float bd = -1.0f; int bj = 0;
#pragma unroll
        for (int i = 0; i < 16; ++i) {
            int j = i * 1024 + t;
            if (d[i] > bd || (d[i] == bd && j < bj)) { bd = d[i]; bj = j; }
        }
#pragma unroll
        for (int off = 16; off > 0; off >>= 1) {
            float od = __shfl_xor(bd, off, 32);
            int   oj = __shfl_xor(bj, off, 32);
            if (od > bd || (od == bd && oj < bj)) { bd = od; bj = oj; }
        }
        if (lane == 0) { sval[wid] = bd; sjdx[wid] = bj; }
        __syncthreads();
        if (t < 32) {
            bd = sval[t]; bj = sjdx[t];
#pragma unroll
            for (int off = 16; off > 0; off >>= 1) {
                float od = __shfl_xor(bd, off, 32);
                int   oj = __shfl_xor(bj, off, 32);
                if (od > bd || (od == bd && oj < bj)) { bd = od; bj = oj; }
            }
            if (t == 0) { swin = bj; sidx[b * QPTS + s] = bj; }
        }
        __syncthreads();
        int wj = swin;
        const float* wp = pc + base0 + (size_t)wj * 35;
        float wx = wp[0], wy = wp[1], wz = wp[2];
#pragma unroll
        for (int i = 0; i < 16; ++i) {
            float dx = px[i] - wx, dy = py[i] - wy, dz = pz[i] - wz;
            float dd = dx * dx + dy * dy + dz * dz;
            d[i] = fminf(d[i], dd);
        }
    }
}

// ---------------------------------------------------------------------------
// Convert feats to f16, compute |p|^2.
// ---------------------------------------------------------------------------
__global__ void prep_feats_kernel(const float* __restrict__ pc,
                                  _Float16* __restrict__ feats16,
                                  float* __restrict__ pnorm, int total) {
    int t = blockIdx.x * blockDim.x + threadIdx.x;
    if (t >= total) return;
    const float* src = pc + (size_t)t * 35 + 3;
    _Float16* dst = feats16 + (size_t)t * FEAT;
    float s = 0.f;
#pragma unroll
    for (int c = 0; c < FEAT; ++c) {
        float v = src[c];
        s += v * v;
        dst[c] = (_Float16)v;
    }
    pnorm[t] = s;
}

// ---------------------------------------------------------------------------
// Gather sampled points: spos -> out[:, :, 0:3], sfeat16, qnorm.
// ---------------------------------------------------------------------------
__global__ void sample_kernel(const float* __restrict__ pc, const int* __restrict__ sidx,
                              const _Float16* __restrict__ feats16,
                              const float* __restrict__ pnorm,
                              _Float16* __restrict__ sfeat16, float* __restrict__ qnorm,
                              float* __restrict__ out) {
    int t = blockIdx.x * blockDim.x + threadIdx.x;
    if (t >= BATCH * QPTS) return;
    int b = t >> 9;
    int idx = sidx[t];
    const float* p = pc + ((size_t)b * NPTS + idx) * 35;
    float* o = out + (size_t)t * 259;
    o[0] = p[0]; o[1] = p[1]; o[2] = p[2];
    const uint4* src = (const uint4*)(feats16 + ((size_t)b * NPTS + idx) * FEAT);
    uint4* dst = (uint4*)(sfeat16 + (size_t)t * FEAT);
    dst[0] = src[0]; dst[1] = src[1]; dst[2] = src[2]; dst[3] = src[3];
    qnorm[t] = pnorm[b * NPTS + idx];
}

// ---------------------------------------------------------------------------
// Ball query, both radii at once. 1 block = (batch b, 16-query tile).
// 4 waves fill d2[16][128] via WMMA, 16 scanner threads select first-K.
// ---------------------------------------------------------------------------
__global__ __launch_bounds__(128) void ball_query_kernel(
    const _Float16* __restrict__ sfeat16, const _Float16* __restrict__ feats16,
    const float* __restrict__ qnorm, const float* __restrict__ pnorm,
    int* __restrict__ bidx1, int* __restrict__ bidx2) {
    __shared__ float d2s[16 * 128];
    __shared__ float qn[16];
    __shared__ int ndone;

    const int b  = blockIdx.x >> 5;
    const int qt = blockIdx.x & 31;
    const int tid = threadIdx.x;
    const int wid = tid >> 5, lane = tid & 31, lm = lane & 15;
    const bool hi = lane >= 16;

    // A fragment (queries): row = qt*16+lm, K halves split by half-wave.
    const _Float16* ab = sfeat16 + ((size_t)b * QPTS + qt * 16 + lm) * FEAT + (hi ? 8 : 0);
    v16h A = pack16(*(const v8h*)ab, *(const v8h*)(ab + 16));

    if (tid < 16) qn[tid] = qnorm[b * QPTS + qt * 16 + tid];
    if (tid == 0) ndone = 0;
    int c1 = 0, c2 = 0;
    bool counted = false;
    const int qrow = b * QPTS + qt * 16 + tid;  // used only when tid<16
    __syncthreads();

    for (int chunk = 0; chunk < NPTS / 128; ++chunk) {
#pragma unroll
        for (int s = 0; s < 2; ++s) {
            int nt = chunk * 8 + wid * 2 + s;
            int j = nt * 16 + lm;
            const _Float16* bp = feats16 + ((size_t)b * NPTS + j) * FEAT + (hi ? 8 : 0);
            v16h Bf = pack16(*(const v8h*)bp, *(const v8h*)(bp + 16));
            v8f c = {};
            c = __builtin_amdgcn_wmma_f32_16x16x32_f16(false, A, false, Bf, (short)0, c,
                                                       false, false);
            float pn = pnorm[b * NPTS + j];
            int nl = (wid * 2 + s) * 16 + lm;
#pragma unroll
            for (int r = 0; r < 8; ++r) {
                int m = r + (hi ? 8 : 0);
                d2s[m * 128 + nl] = qn[m] + pn - 2.0f * c[r];
            }
        }
        __syncthreads();
        if (tid < 16) {
            for (int k = 0; k < 128; ++k) {
                float d2 = d2s[tid * 128 + k];
                int j = chunk * 128 + k;
                if (d2 < R1SQ && c1 < NS1) bidx1[(size_t)qrow * NS1 + c1++] = j;
                if (d2 < R2SQ && c2 < NS2) bidx2[(size_t)qrow * NS2 + c2++] = j;
            }
            if (!counted && c1 >= NS1 && c2 >= NS2) { counted = true; atomicAdd(&ndone, 1); }
        }
        __syncthreads();
        if (ndone == 16) break;
    }
    if (tid < 16) {
        for (int k = c1; k < NS1; ++k) bidx1[(size_t)qrow * NS1 + k] = -1;
        for (int k = c2; k < NS2; ++k) bidx2[(size_t)qrow * NS2 + k] = -1;
    }
}

// ---------------------------------------------------------------------------
// Weights -> f16, BN folded into per-channel scale/shift.
// ss layout: [s1(64) h1(64) s2(64) h2(64) s3(128) h3(128)]
// ---------------------------------------------------------------------------
__global__ void prep_params_kernel(
    const float* w1, const float* b1, const float* g1, const float* be1,
    const float* m1, const float* v1, const float* w2, const float* b2,
    const float* g2, const float* be2, const float* m2, const float* v2,
    const float* w3, const float* b3, const float* g3, const float* be3,
    const float* m3, const float* v3, _Float16* w16, float* ss) {
    int t = threadIdx.x;
    for (int i = t; i < 64 * 32; i += 256) w16[i] = (_Float16)w1[i];
    for (int i = t; i < 64 * 64; i += 256) w16[2048 + i] = (_Float16)w2[i];
    for (int i = t; i < 128 * 64; i += 256) w16[6144 + i] = (_Float16)w3[i];
    if (t < 64) {
        float inv = g1[t] / sqrtf(v1[t] + 1e-5f);
        ss[t] = inv;
        ss[64 + t] = b1[t] * inv + be1[t] - m1[t] * inv;
        float inv2 = g2[t] / sqrtf(v2[t] + 1e-5f);
        ss[128 + t] = inv2;
        ss[192 + t] = b2[t] * inv2 + be2[t] - m2[t] * inv2;
    }
    if (t < 128) {
        float inv = g3[t] / sqrtf(v3[t] + 1e-5f);
        ss[256 + t] = inv;
        ss[384 + t] = b3[t] * inv + be3[t] - m3[t] * inv;
    }
}

// ---------------------------------------------------------------------------
// Zero workspace (for zero-padded borders & invalid gather slots).
// ---------------------------------------------------------------------------
__global__ void zero_kernel(uint4* __restrict__ p, size_t n16) {
    uint4 z; z.x = z.y = z.z = z.w = 0u;
    for (size_t i = blockIdx.x * (size_t)blockDim.x + threadIdx.x; i < n16;
         i += (size_t)gridDim.x * blockDim.x)
        p[i] = z;
}

// ---------------------------------------------------------------------------
// Masked gather: grp -> zero-padded channel-last f16 buffer (514, ns+2, 32).
// ---------------------------------------------------------------------------
__global__ void gather_grp_kernel(const int* __restrict__ bidx,
                                  const _Float16* __restrict__ feats16,
                                  _Float16* __restrict__ in1, int ns, int total) {
    int t = blockIdx.x * blockDim.x + threadIdx.x;
    if (t >= total) return;
    int s = t % ns, q = (t / ns) & 511, b = t / (ns * QPTS);
    int idx = bidx[t];
    if (idx < 0) return;  // buffer pre-zeroed
    const uint4* src = (const uint4*)(feats16 + ((size_t)b * NPTS + idx) * FEAT);
    uint4* dst = (uint4*)(in1 + (((size_t)b * 514 + (1 + q)) * (ns + 2) + (1 + s)) * FEAT);
    dst[0] = src[0]; dst[1] = src[1]; dst[2] = src[2]; dst[3] = src[3];
}

// ---------------------------------------------------------------------------
// 1x1 conv + BN + ReLU as GEMM. Input buffer IS the padded tensor; output
// written into interior of the next padded buffer.
// Block = 8 waves = 2 M-tiles x 4 N-tiles. The 32-pixel x CIN A-block is
// staged global -> LDS via global_load_async_to_lds_b128 (ASYNCcnt), then
// each wave builds WMMA A fragments from LDS (ds_load_b128). Weights are
// tiny/L2-hot and loaded directly.
// ---------------------------------------------------------------------------
template <int CIN, int COUT, int WS>
__global__ __launch_bounds__(256) void conv_bn_relu_kernel(
    const _Float16* __restrict__ in, _Float16* __restrict__ out,
    const _Float16* __restrict__ w16, const float* __restrict__ scale,
    const float* __restrict__ shift, int Hs) {
    const int M = Hs * WS;
    const int ntiles = COUT / 16;       // 4
    const int ROWS = (8 / ntiles) * 16; // 32 pixels per block
    __shared__ __align__(16) _Float16 As[ROWS * CIN];

    const int mblocks = (M + ROWS - 1) / ROWS;
    const int b  = blockIdx.x / mblocks;
    const int mb = blockIdx.x % mblocks;
    const int p0 = mb * ROWS;
    const int t = threadIdx.x;

    // ---- async stage: A block (ROWS x CIN f16, contiguous) -> LDS ----
    const char* gsrc = (const char*)(in + ((size_t)b * M + p0) * CIN);
    const uint32_t lds0 = (uint32_t)(uintptr_t)(&As[0]);
    const int nchunks = ROWS * CIN * 2 / 16;  // 128 (CIN=32) or 256 (CIN=64)
    for (int c = t; c < nchunks; c += 256) {
        int pix = (c * 16) / (CIN * 2);
        // tail M-tiles: clamp OOB chunks to row 0 (content unused, store-guarded)
        size_t goff = (p0 + pix < M) ? (size_t)c * 16 : 0;
        uint32_t la = lds0 + (uint32_t)c * 16;
        asm volatile("global_load_async_to_lds_b128 %0, %1, off"
                     :: "v"(la), "v"(gsrc + goff)
                     : "memory");
    }
    asm volatile("s_wait_asynccnt 0" ::: "memory");
    __syncthreads();

    const int wid = t >> 5, lane = t & 31, lm = lane & 15;
    const bool hi = lane >= 16;
    const int mtl = wid / ntiles, nt = wid % ntiles;

    const int o = nt * 16 + lm;
    v16h Bf[CIN / 32];
#pragma unroll
    for (int k = 0; k < CIN / 32; ++k) {
        const _Float16* wb = w16 + (size_t)o * CIN + k * 32 + (hi ? 8 : 0);
        Bf[k] = pack16(*(const v8h*)wb, *(const v8h*)(wb + 16));
    }
    v8f c = {};
#pragma unroll
    for (int k = 0; k < CIN / 32; ++k) {
        const _Float16* ar = &As[(mtl * 16 + lm) * CIN + k * 32 + (hi ? 8 : 0)];
        v16h A = pack16(*(const v8h*)ar, *(const v8h*)(ar + 16));
        c = __builtin_amdgcn_wmma_f32_16x16x32_f16(false, A, false, Bf[k], (short)0, c,
                                                   false, false);
    }
    const float sc = scale[o], sh = shift[o];
    _Float16* ob = out + (size_t)b * (Hs + 2) * (WS + 2) * COUT;
#pragma unroll
    for (int r = 0; r < 8; ++r) {
        int m = p0 + mtl * 16 + r + (hi ? 8 : 0);
        if (m < M) {
            int h = m / WS, w = m % WS;
            float v = c[r] * sc + sh;
            v = v > 0.f ? v : 0.f;
            ob[((size_t)(h + 1) * (WS + 2) + (w + 1)) * COUT + o] = (_Float16)v;
        }
    }
}

// ---------------------------------------------------------------------------
// conv3 (64->128) fused with max over W (BN scale>0 commutes with max).
// 1 wave = (batch, 16 rows of H=518, 16 out channels), loops W.
// ---------------------------------------------------------------------------
template <int W3>
__global__ __launch_bounds__(256) void conv3_wmax_kernel(
    const _Float16* __restrict__ in, float* __restrict__ hmax,
    const _Float16* __restrict__ w16, const float* __restrict__ scale,
    const float* __restrict__ shift, int totalWaves) {
    const int H3 = 518;
    int wt = blockIdx.x * 8 + (threadIdx.x >> 5);
    if (wt >= totalWaves) return;
    const int lane = threadIdx.x & 31, lm = lane & 15;
    const bool hi = lane >= 16;
    const int ot = wt & 7;
    const int r1 = wt >> 3;
    const int ht = r1 % 33, b = r1 / 33;

    const int o = ot * 16 + lm;
    const _Float16* wb = w16 + (size_t)o * 64;
    v16h B0 = pack16(*(const v8h*)(wb + (hi ? 8 : 0)), *(const v8h*)(wb + (hi ? 8 : 0) + 16));
    v16h B1 = pack16(*(const v8h*)(wb + 32 + (hi ? 8 : 0)),
                     *(const v8h*)(wb + 32 + (hi ? 8 : 0) + 16));

    const int h = ht * 16 + lm;
    const bool hv = h < H3;
    const _Float16* ab0 = in + ((size_t)b * H3 + (hv ? h : 0)) * W3 * 64;
    float acc[8];
#pragma unroll
    for (int r = 0; r < 8; ++r) acc[r] = -3.4e38f;

    for (int w = 0; w < W3; ++w) {
        const _Float16* ap = ab0 + (size_t)w * 64;
        v8h a0l = {}, a0h = {}, a1l = {}, a1h = {};
        if (hv) {
            a0l = *(const v8h*)(ap + (hi ? 8 : 0));
            a0h = *(const v8h*)(ap + 16 + (hi ? 8 : 0));
            a1l = *(const v8h*)(ap + 32 + (hi ? 8 : 0));
            a1h = *(const v8h*)(ap + 48 + (hi ? 8 : 0));
        }
        v16h A0 = pack16(a0l, a0h), A1 = pack16(a1l, a1h);
        v8f c = {};
        c = __builtin_amdgcn_wmma_f32_16x16x32_f16(false, A0, false, B0, (short)0, c,
                                                   false, false);
        c = __builtin_amdgcn_wmma_f32_16x16x32_f16(false, A1, false, B1, (short)0, c,
                                                   false, false);
#pragma unroll
        for (int r = 0; r < 8; ++r) acc[r] = fmaxf(acc[r], c[r]);
    }
    const float sc = scale[o], sh = shift[o];
#pragma unroll
    for (int r = 0; r < 8; ++r) {
        int hh = ht * 16 + r + (hi ? 8 : 0);
        if (hh < H3) {
            float v = fmaxf(acc[r] * sc + sh, 0.f);
            hmax[((size_t)b * H3 + hh) * 128 + o] = v;
        }
    }
}

// ---------------------------------------------------------------------------
// Adaptive max-pool 518 -> 512 rows, write into output feature columns.
// ---------------------------------------------------------------------------
__global__ void adaptive_pool_kernel(const float* __restrict__ hmax,
                                     float* __restrict__ out, int colOff) {
    int t = blockIdx.x * blockDim.x + threadIdx.x;
    if (t >= BATCH * QPTS * 128) return;
    int o = t & 127, i = (t >> 7) & 511, b = t >> 16;
    int start = (i * 518) >> 9;
    int end = ((i + 1) * 518 + 511) >> 9;
    float v = -3.4e38f;
    for (int h = start; h < end; ++h) v = fmaxf(v, hmax[((size_t)b * 518 + h) * 128 + o]);
    out[((size_t)(b * QPTS + i)) * 259 + 3 + colOff + o] = v;
}

// ---------------------------------------------------------------------------
extern "C" void kernel_launch(void* const* d_in, const int* in_sizes, int n_in,
                              void* d_out, int out_size, void* d_ws, size_t ws_size,
                              hipStream_t stream) {
    (void)in_sizes; (void)n_in; (void)out_size; (void)ws_size;
    const float* pc = (const float*)d_in[0];
    const float *w1 = (const float*)d_in[1],  *b1 = (const float*)d_in[2],
                *g1 = (const float*)d_in[3],  *be1 = (const float*)d_in[4],
                *m1 = (const float*)d_in[5],  *v1 = (const float*)d_in[6];
    const float *w2 = (const float*)d_in[7],  *b2 = (const float*)d_in[8],
                *g2 = (const float*)d_in[9],  *be2 = (const float*)d_in[10],
                *m2 = (const float*)d_in[11], *v2 = (const float*)d_in[12];
    const float *w3 = (const float*)d_in[13], *b3 = (const float*)d_in[14],
                *g3 = (const float*)d_in[15], *be3 = (const float*)d_in[16],
                *m3 = (const float*)d_in[17], *v3 = (const float*)d_in[18];
    float* out = (float*)d_out;
    char* ws = (char*)d_ws;

    size_t off = 0;
    auto alloc = [&](size_t bytes) {
        size_t o = off;
        off = (off + bytes + 255) & ~(size_t)255;
        return o;
    };
    _Float16* feats16 = (_Float16*)(ws + alloc((size_t)BATCH * NPTS * FEAT * 2));
    float*    pnorm   = (float*)(ws + alloc((size_t)BATCH * NPTS * 4));
    _Float16* sfeat16 = (_Float16*)(ws + alloc((size_t)BATCH * QPTS * FEAT * 2));
    float*    qnorm   = (float*)(ws + alloc((size_t)BATCH * QPTS * 4));
    int*      sidx    = (int*)(ws + alloc((size_t)BATCH * QPTS * 4));
    int*      bidx1   = (int*)(ws + alloc((size_t)BATCH * QPTS * NS1 * 4));
    int*      bidx2   = (int*)(ws + alloc((size_t)BATCH * QPTS * NS2 * 4));
    _Float16* w16     = (_Float16*)(ws + alloc(14336 * 2));
    float*    ss      = (float*)(ws + alloc(512 * 4));
    // Activation region (contiguous, zeroed per branch). Sized for ns=64.
    size_t in1Bytes  = (size_t)BATCH * 514 * 66 * 32 * 2;   // 17,369,088
    size_t act1Bytes = (size_t)BATCH * 516 * 68 * 64 * 2;   // 35,930,112
    size_t act2Bytes = (size_t)BATCH * 518 * 70 * 64 * 2;   // 37,130,240
    _Float16* in1  = (_Float16*)(ws + alloc(in1Bytes));
    _Float16* act1 = (_Float16*)(ws + alloc(act1Bytes));
    _Float16* act2 = (_Float16*)(ws + alloc(act2Bytes));
    float*    hmax = (float*)(ws + alloc((size_t)BATCH * 518 * 128 * 4));
    const size_t zeroN16 = (in1Bytes + act1Bytes + act2Bytes) / 16;

    // ---- shared preprocessing ----
    prep_params_kernel<<<1, 256, 0, stream>>>(w1, b1, g1, be1, m1, v1, w2, b2, g2, be2,
                                              m2, v2, w3, b3, g3, be3, m3, v3, w16, ss);
    prep_feats_kernel<<<(BATCH * NPTS + 255) / 256, 256, 0, stream>>>(pc, feats16, pnorm,
                                                                      BATCH * NPTS);
    fps_kernel<<<BATCH, 1024, 0, stream>>>(pc, sidx);
    sample_kernel<<<(BATCH * QPTS + 255) / 256, 256, 0, stream>>>(pc, sidx, feats16, pnorm,
                                                                  sfeat16, qnorm, out);
    ball_query_kernel<<<BATCH * 32, 128, 0, stream>>>(sfeat16, feats16, qnorm, pnorm,
                                                      bidx1, bidx2);

    // ---- branch 1: radius 6, ns=32 (W dims 34/36/38) ----
    {
        zero_kernel<<<2048, 256, 0, stream>>>((uint4*)in1, zeroN16);
        gather_grp_kernel<<<(BATCH * QPTS * NS1 + 255) / 256, 256, 0, stream>>>(
            bidx1, feats16, in1, NS1, BATCH * QPTS * NS1);
        int m1b = (514 * 34 + 31) / 32;  // 547
        conv_bn_relu_kernel<32, 64, 34><<<BATCH * m1b, 256, 0, stream>>>(in1, act1, w16,
                                                                         ss, ss + 64, 514);
        int m2b = (516 * 36 + 31) / 32;  // 581
        conv_bn_relu_kernel<64, 64, 36><<<BATCH * m2b, 256, 0, stream>>>(
            act1, act2, w16 + 2048, ss + 128, ss + 192, 516);
        conv3_wmax_kernel<38><<<264, 256, 0, stream>>>(act2, hmax, w16 + 6144, ss + 256,
                                                       ss + 384, 2112);
        adaptive_pool_kernel<<<(BATCH * QPTS * 128 + 255) / 256, 256, 0, stream>>>(hmax,
                                                                                   out, 0);
    }

    // ---- branch 2: radius 8, ns=64 (W dims 66/68/70) ----
    {
        zero_kernel<<<2048, 256, 0, stream>>>((uint4*)in1, zeroN16);
        gather_grp_kernel<<<(BATCH * QPTS * NS2 + 255) / 256, 256, 0, stream>>>(
            bidx2, feats16, in1, NS2, BATCH * QPTS * NS2);
        int m1b = (514 * 66 + 31) / 32;  // 1061
        conv_bn_relu_kernel<32, 64, 66><<<BATCH * m1b, 256, 0, stream>>>(in1, act1, w16,
                                                                         ss, ss + 64, 514);
        int m2b = (516 * 68 + 31) / 32;  // 1097
        conv_bn_relu_kernel<64, 64, 68><<<BATCH * m2b, 256, 0, stream>>>(
            act1, act2, w16 + 2048, ss + 128, ss + 192, 516);
        conv3_wmax_kernel<70><<<264, 256, 0, stream>>>(act2, hmax, w16 + 6144, ss + 256,
                                                       ss + 384, 2112);
        adaptive_pool_kernel<<<(BATCH * QPTS * 128 + 255) / 256, 256, 0, stream>>>(hmax,
                                                                                   out, 128);
    }
}
